// MLA_2680059593175
// MI455X (gfx1250) — matrix operand order repaired and checked by reading, not compiled, not run on hardware
//
#include <hip/hip_runtime.h>

#define B_DIM 2
#define S_LEN 2048
#define HEADS 16
#define DH 192
#define DV 128
#define ROWS 4096   // B*S

typedef __attribute__((ext_vector_type(16))) __bf16 bf16x16;
typedef __attribute__((ext_vector_type(8)))  float  f32x8;
typedef int v4i __attribute__((vector_size(16)));
typedef unsigned int uv4 __attribute__((vector_size(16)));

union BF16Frag { bf16x16 v; uv4 u[2]; };

#ifndef __has_builtin
#define __has_builtin(x) 0
#endif
#if __has_builtin(__builtin_amdgcn_global_load_async_to_lds_b128)
#define HAVE_ASYNC 1
#else
#define HAVE_ASYNC 0
#endif

#define AS1 __attribute__((address_space(1)))
#define AS3 __attribute__((address_space(3)))

// force GLOBAL (not FLAT) vector loads/stores; uv4 is a native vector (no ctor)
static __device__ __forceinline__ uv4 gload16(const void* p) {
  return *(const AS1 uv4*)(unsigned long long)p;
}
static __device__ __forceinline__ void gstore_f32(void* p, float v) {
  *(AS1 float*)(unsigned long long)p = v;
}
static __device__ __forceinline__ void gstore_bf16(void* p, __bf16 v) {
  *(AS1 __bf16*)(unsigned long long)p = v;
}

#if HAVE_ASYNC
// CDNA5 async global->LDS copy (ASYNCcnt). LDS offset = low 32 bits of generic ptr.
static __device__ __forceinline__ void async_cp16(const void* g, void* l) {
  __builtin_amdgcn_global_load_async_to_lds_b128(
      (AS1 v4i*)(unsigned long long)g,
      (AS3 v4i*)(unsigned)(unsigned long long)l, 0, 0);
}
static __device__ __forceinline__ void async_wait0() {
#if __has_builtin(__builtin_amdgcn_s_wait_asynccnt)
  __builtin_amdgcn_s_wait_asynccnt(0);
#else
  asm volatile("s_wait_asynccnt 0x0" ::: "memory");
#endif
}
#endif

static __device__ __forceinline__ f32x8 wmma_bf16(const BF16Frag& a, const BF16Frag& b, f32x8 c) {
  return __builtin_amdgcn_wmma_f32_16x16x32_bf16(false, a.v, false, b.v, (short)0, c, false, false);
}

static __device__ __forceinline__ float halfmax16(float v) {
  #pragma unroll
  for (int off = 1; off < 16; off <<= 1) v = fmaxf(v, __shfl_xor(v, off, 32));
  return v;
}
static __device__ __forceinline__ float halfsum16(float v) {
  #pragma unroll
  for (int off = 1; off < 16; off <<= 1) v += __shfl_xor(v, off, 32);
  return v;
}

// ---------------------------------------------------------------------------
__global__ __launch_bounds__(256)
void k_convert(const float* __restrict__ src, __bf16* __restrict__ dst, long long n) {
  for (long long i = (long long)blockIdx.x * 256 + threadIdx.x; i < n; i += (long long)gridDim.x * 256)
    dst[i] = (__bf16)src[i];
}

// transpose fp32 [K,N] -> bf16 [N,K]
__global__ __launch_bounds__(256)
void k_transpose(const float* __restrict__ src, __bf16* __restrict__ dst, int K, int N) {
  long long total = (long long)K * N;
  for (long long idx = (long long)blockIdx.x * 256 + threadIdx.x; idx < total; idx += (long long)gridDim.x * 256) {
    int n = (int)(idx % N);
    long long k = idx / N;
    dst[(long long)n * K + k] = (__bf16)src[idx];
  }
}

// ---------------------------------------------------------------------------
// bf16 WMMA GEMM: C[M,N](f32) = A[M,K] * Bt[N,K]^T. 8 waves, 128x128 tile,
// K-step 32, double-buffered LDS, async global->LDS when available.
__global__ __launch_bounds__(256)
void k_gemm(const __bf16* __restrict__ A, const __bf16* __restrict__ Bt,
            float* __restrict__ C, int M, int N, int K) {
  __shared__ __bf16 As[2][128][40];
  __shared__ __bf16 Bs[2][128][40];
  const int tid  = threadIdx.x;
  const int wave = tid >> 5;
  const int lane = tid & 31;
  const int l16  = lane & 15;
  const int half = lane >> 4;
  const int m0 = blockIdx.y * 128;
  const int n0 = blockIdx.x * 128;
  const int wm = (wave & 3) * 32;
  const int wn = (wave >> 2) * 64;

  f32x8 acc[2][4] = {};

  const int ldRow = tid >> 1;
  const int ldOff = (tid & 1) * 16;
  const __bf16* aSrc = A + (size_t)(m0 + ldRow) * K + ldOff;
  const int bn  = n0 + ldRow;
  const int bnc = (bn < N) ? bn : (N - 1);      // clamp: garbage cols never stored
  const __bf16* bSrc = Bt + (size_t)bnc * K + ldOff;

  auto compute = [&](int cur) {
    BF16Frag af[2], bfr[4];
    #pragma unroll
    for (int mt = 0; mt < 2; ++mt) {
      const __bf16* r = &As[cur][wm + mt * 16 + l16][0];
      af[mt].u[0] = *(const uv4*)(r + half * 8);        // K = half*8 + 0..7
      af[mt].u[1] = *(const uv4*)(r + 16 + half * 8);   // K = 16 + half*8 + 0..7
    }
    #pragma unroll
    for (int nt = 0; nt < 4; ++nt) {
      const __bf16* r = &Bs[cur][wn + nt * 16 + l16][0];
      bfr[nt].u[0] = *(const uv4*)(r + half * 8);       // K = half*8 + 0..7
      bfr[nt].u[1] = *(const uv4*)(r + 16 + half * 8);  // K = 16 + half*8 + 0..7
    }
    #pragma unroll
    for (int mt = 0; mt < 2; ++mt)
      #pragma unroll
      for (int nt = 0; nt < 4; ++nt)
        acc[mt][nt] = wmma_bf16(af[mt], bfr[nt], acc[mt][nt]);
  };

#if HAVE_ASYNC
  auto stage = [&](int k0, int buf) {
    async_cp16(aSrc + k0,     &As[buf][ldRow][ldOff]);
    async_cp16(aSrc + k0 + 8, &As[buf][ldRow][ldOff + 8]);
    async_cp16(bSrc + k0,     &Bs[buf][ldRow][ldOff]);
    async_cp16(bSrc + k0 + 8, &Bs[buf][ldRow][ldOff + 8]);
  };
  stage(0, 0);
  async_wait0();
  __syncthreads();
  for (int k0 = 0; k0 < K; k0 += 32) {
    const int cur = (k0 >> 5) & 1;
    if (k0 + 32 < K) stage(k0 + 32, cur ^ 1);
    compute(cur);
    async_wait0();
    __syncthreads();
  }
#else
  uv4 a0 = gload16(aSrc),     a1 = gload16(aSrc + 8);
  uv4 b0 = gload16(bSrc),     b1 = gload16(bSrc + 8);
  *(uv4*)&As[0][ldRow][ldOff]     = a0;
  *(uv4*)&As[0][ldRow][ldOff + 8] = a1;
  *(uv4*)&Bs[0][ldRow][ldOff]     = b0;
  *(uv4*)&Bs[0][ldRow][ldOff + 8] = b1;
  __syncthreads();
  for (int k0 = 0; k0 < K; k0 += 32) {
    const int cur  = (k0 >> 5) & 1;
    const bool more = (k0 + 32) < K;
    if (more) {
      a0 = gload16(aSrc + k0 + 32); a1 = gload16(aSrc + k0 + 40);
      b0 = gload16(bSrc + k0 + 32); b1 = gload16(bSrc + k0 + 40);
      if (k0 + 64 < K) {
        __builtin_prefetch(aSrc + k0 + 64, 0, 3);
        __builtin_prefetch(bSrc + k0 + 64, 0, 3);
      }
    }
    compute(cur);
    if (more) {
      const int nxt = cur ^ 1;
      *(uv4*)&As[nxt][ldRow][ldOff]     = a0;
      *(uv4*)&As[nxt][ldRow][ldOff + 8] = a1;
      *(uv4*)&Bs[nxt][ldRow][ldOff]     = b0;
      *(uv4*)&Bs[nxt][ldRow][ldOff + 8] = b1;
    }
    __syncthreads();
  }
#endif

  #pragma unroll
  for (int mt = 0; mt < 2; ++mt)
    #pragma unroll
    for (int nt = 0; nt < 4; ++nt) {
      const int col = n0 + wn + nt * 16 + l16;
      if (col < N) {
        #pragma unroll
        for (int r = 0; r < 8; ++r) {
          const int row = m0 + wm + mt * 16 + r + 8 * half;
          gstore_f32(&C[(size_t)row * N + col], acc[mt][nt][r]);
        }
      }
    }
}

// ---------------------------------------------------------------------------
__global__ __launch_bounds__(256)
void k_rmsnorm_qa(const float* __restrict__ X, const float* __restrict__ g,
                  __bf16* __restrict__ Y) {
  const int row = blockIdx.x;
  const int tid = threadIdx.x;
  const float* x = X + (size_t)row * 1536;
  float v[6], ss = 0.f;
  #pragma unroll
  for (int i = 0; i < 6; ++i) { v[i] = x[tid + i * 256]; ss += v[i] * v[i]; }
  __shared__ float red[8];
  #pragma unroll
  for (int off = 1; off < 32; off <<= 1) ss += __shfl_xor(ss, off, 32);
  if ((tid & 31) == 0) red[tid >> 5] = ss;
  __syncthreads();
  float tot = red[0];
  #pragma unroll
  for (int i = 1; i < 8; ++i) tot += red[i];
  const float r = rsqrtf(tot * (1.f / 1536.f) + 1e-6f);
  #pragma unroll
  for (int i = 0; i < 6; ++i)
    Y[(size_t)row * 1536 + tid + i * 256] = (__bf16)(v[i] * r * g[tid + i * 256]);
}

__global__ __launch_bounds__(256)
void k_rmsnorm_kva(const float* __restrict__ X, const float* __restrict__ g,
                   __bf16* __restrict__ Yn, float* __restrict__ Kpe) {
  const int row = blockIdx.x;
  const int tid = threadIdx.x;
  const float* x = X + (size_t)row * 576;
  float v0 = x[tid], v1 = x[tid + 256];
  float ss = v0 * v0 + v1 * v1;
  __shared__ float red[8];
  #pragma unroll
  for (int off = 1; off < 32; off <<= 1) ss += __shfl_xor(ss, off, 32);
  if ((tid & 31) == 0) red[tid >> 5] = ss;
  __syncthreads();
  float tot = red[0];
  #pragma unroll
  for (int i = 1; i < 8; ++i) tot += red[i];
  const float r = rsqrtf(tot * (1.f / 512.f) + 1e-6f);
  Yn[(size_t)row * 512 + tid]       = (__bf16)(v0 * r * g[tid]);
  Yn[(size_t)row * 512 + tid + 256] = (__bf16)(v1 * r * g[tid + 256]);
  if (tid < 64) {
    const int s = row & (S_LEN - 1);
    const int j = tid, jj = j & 31;
    const float inv = __powf(10000.f, -(float)jj * (1.f / 32.f));
    const float ang = (float)s * inv;
    const float xv = x[512 + j];
    const float xp = (j < 32) ? -x[512 + j + 32] : x[512 + j - 32];
    Kpe[(size_t)row * 64 + j] = xv * cosf(ang) + xp * sinf(ang);
  }
}

// ---------------------------------------------------------------------------
__global__ __launch_bounds__(256)
void k_rope_q(const float* __restrict__ Qr, __bf16* __restrict__ Qf) {
  const int total = ROWS * HEADS * DH;
  for (int idx = blockIdx.x * 256 + threadIdx.x; idx < total; idx += gridDim.x * 256) {
    int d = idx % DH;
    int t = idx / DH;
    int h = t & (HEADS - 1); t >>= 4;
    int s = t & (S_LEN - 1);
    int b = t >> 11;
    const float* base = Qr + ((size_t)((b * S_LEN + s) * HEADS + h)) * DH;
    float val = base[d];
    if (d >= 128) {
      int j = d - 128, jj = j & 31;
      float inv = __powf(10000.f, -(float)jj * (1.f / 32.f));
      float ang = (float)s * inv;
      float xp = (j < 32) ? -base[d + 32] : base[d - 32];
      val = val * cosf(ang) + xp * sinf(ang);
    }
    Qf[((size_t)(b * HEADS + h) * S_LEN + s) * DH + d] = (__bf16)val;
  }
}

__global__ __launch_bounds__(256)
void k_rearrange_kv(const float* __restrict__ KV, const float* __restrict__ Kpe,
                    __bf16* __restrict__ Kf, __bf16* __restrict__ Vt) {
  const int T1 = ROWS * HEADS * 256;
  const int T2 = B_DIM * HEADS * S_LEN * 64;
  for (int idx = blockIdx.x * 256 + threadIdx.x; idx < T1 + T2; idx += gridDim.x * 256) {
    if (idx < T1) {
      int d = idx & 255;
      int t = idx >> 8;
      int h = t & (HEADS - 1); t >>= 4;
      int s = t & (S_LEN - 1);
      int b = t >> 11;
      int bh = b * HEADS + h;
      float val = KV[idx];
      if (d < 128)
        Kf[((size_t)bh * S_LEN + s) * DH + d] = (__bf16)val;
      else
        Vt[((size_t)bh * DV + (d - 128)) * S_LEN + s] = (__bf16)val;
    } else {
      int i2 = idx - T1;
      int j = i2 & 63;
      int t = i2 >> 6;
      int s = t & (S_LEN - 1);
      int bh = t >> 11;
      Kf[((size_t)bh * S_LEN + s) * DH + 128 + j] =
          (__bf16)Kpe[((size_t)(bh >> 4) * S_LEN + s) * 64 + j];
    }
  }
}

// ---------------------------------------------------------------------------
// Causal flash attention. grid=(16 q-blocks, 32 bh). 8 waves x 16 q rows.
__global__ __launch_bounds__(256)
void k_attention(const __bf16* __restrict__ Q, const __bf16* __restrict__ Kf,
                 const __bf16* __restrict__ Vt, __bf16* __restrict__ Ctx) {
  const int bh  = blockIdx.y;
  const int b   = bh >> 4, h = bh & 15;
  const int qb  = blockIdx.x;
  const int tid = threadIdx.x;
  const int wave = tid >> 5;
  const int lane = tid & 31;
  const int l16  = lane & 15;
  const int half = lane >> 4;

  __shared__ __bf16 Ks[64][DH + 8];   // 64 x 200
  __shared__ __bf16 Vs[DV][64 + 8];   // 128 x 72 (d-major)
  __shared__ __bf16 Ps[8][16][72];    // per-wave P tile

  const __bf16* qptr = Q  + (size_t)bh * S_LEN * DH;
  const __bf16* kptr = Kf + (size_t)bh * S_LEN * DH;
  const __bf16* vptr = Vt + (size_t)bh * DV * S_LEN;

  const int qRow0 = qb * 128 + wave * 16;

  BF16Frag qf[6];
  {
    const __bf16* qr = qptr + (size_t)(qRow0 + l16) * DH;
    #pragma unroll
    for (int c = 0; c < 6; ++c) {
      qf[c].u[0] = gload16(qr + c * 32 + half * 8);
      qf[c].u[1] = gload16(qr + c * 32 + 16 + half * 8);
    }
  }

  f32x8 o[8] = {};
  float m_r[8], l_r[8];
  #pragma unroll
  for (int r = 0; r < 8; ++r) { m_r[r] = -3.0e30f; l_r[r] = 0.f; }

  const float sc = 0.072168784f; // 1/sqrt(192)
  const int nkb = qb * 2 + 2;

  const int krow = tid >> 2, kq4 = (tid & 3) * 48;
  const int vd = tid >> 1, vhr = (tid & 1) * 32;

  for (int kb = 0; kb < nkb; ++kb) {
    __syncthreads();
#if HAVE_ASYNC
    {
      const __bf16* src = kptr + (size_t)(kb * 64 + krow) * DH + kq4;
      __bf16* dst = &Ks[krow][kq4];
      #pragma unroll
      for (int i = 0; i < 6; ++i) async_cp16(src + i * 8, dst + i * 8);
      const __bf16* vsrc = vptr + (size_t)vd * S_LEN + kb * 64 + vhr;
      __bf16* vdst = &Vs[vd][vhr];
      #pragma unroll
      for (int i = 0; i < 4; ++i) async_cp16(vsrc + i * 8, vdst + i * 8);
    }
    async_wait0();
#else
    {
      const __bf16* src = kptr + (size_t)(kb * 64 + krow) * DH + kq4;
      __bf16* dst = &Ks[krow][kq4];
      #pragma unroll
      for (int i = 0; i < 6; ++i)
        *(uv4*)(dst + i * 8) = gload16(src + i * 8);
      const __bf16* vsrc = vptr + (size_t)vd * S_LEN + kb * 64 + vhr;
      __bf16* vdst = &Vs[vd][vhr];
      #pragma unroll
      for (int i = 0; i < 4; ++i)
        *(uv4*)(vdst + i * 8) = gload16(vsrc + i * 8);
    }
#endif
    __syncthreads();

    // S = Q (16x192) * K^T -> 4 tiles of 16x16
    f32x8 s[4];
    #pragma unroll
    for (int nt = 0; nt < 4; ++nt) {
      f32x8 a = {};
      #pragma unroll
      for (int c = 0; c < 6; ++c) {
        BF16Frag bk;
        const __bf16* kr = &Ks[nt * 16 + l16][c * 32 + half * 8];
        bk.u[0] = *(const uv4*)(kr);
        bk.u[1] = *(const uv4*)(kr + 16);
        a = wmma_bf16(qf[c], bk, a);
      }
      s[nt] = a;
    }

    // online softmax per row (row M = r + 8*half)
    #pragma unroll
    for (int r = 0; r < 8; ++r) {
      const int qrow = qRow0 + r + 8 * half;
      float mx = -3.0e30f;
      #pragma unroll
      for (int nt = 0; nt < 4; ++nt) {
        const int key = kb * 64 + nt * 16 + l16;
        float v = s[nt][r] * sc;
        v = (key <= qrow) ? v : -3.0e30f;
        s[nt][r] = v;
        mx = fmaxf(mx, v);
      }
      mx = halfmax16(mx);
      const float mnew = fmaxf(m_r[r], mx);
      const float corr = __expf(m_r[r] - mnew);
      m_r[r] = mnew;
      float rs = 0.f;
      #pragma unroll
      for (int nt = 0; nt < 4; ++nt) {
        float p = __expf(s[nt][r] - mnew);
        s[nt][r] = p;
        rs += p;
      }
      rs = halfsum16(rs);
      l_r[r] = l_r[r] * corr + rs;
      #pragma unroll
      for (int vt = 0; vt < 8; ++vt) o[vt][r] *= corr;
    }

    // P (C-layout) -> LDS -> reload as A-frags (per-wave region)
    #pragma unroll
    for (int nt = 0; nt < 4; ++nt)
      #pragma unroll
      for (int r = 0; r < 8; ++r)
        Ps[wave][r + 8 * half][nt * 16 + l16] = (__bf16)s[nt][r];

    BF16Frag pf[2];
    {
      const __bf16* pr = &Ps[wave][l16][0];
      #pragma unroll
      for (int kc = 0; kc < 2; ++kc) {
        pf[kc].u[0] = *(const uv4*)(pr + kc * 32 + half * 8);
        pf[kc].u[1] = *(const uv4*)(pr + kc * 32 + 16 + half * 8);
      }
    }

    // O += P (16x64) * V (64x128)
    #pragma unroll
    for (int vt = 0; vt < 8; ++vt) {
      f32x8 a = o[vt];
      #pragma unroll
      for (int kc = 0; kc < 2; ++kc) {
        BF16Frag bv;
        const __bf16* vr = &Vs[vt * 16 + l16][kc * 32 + half * 8];
        bv.u[0] = *(const uv4*)(vr);
        bv.u[1] = *(const uv4*)(vr + 16);
        a = wmma_bf16(pf[kc], bv, a);
      }
      o[vt] = a;
    }
  }

  #pragma unroll
  for (int r = 0; r < 8; ++r) {
    const float inv = 1.0f / l_r[r];
    const int qrow = qRow0 + r + 8 * half;
    __bf16* dst = Ctx + ((size_t)(b * S_LEN + qrow)) * (HEADS * DV) + h * DV;
    #pragma unroll
    for (int vt = 0; vt < 8; ++vt)
      gstore_bf16(dst + vt * 16 + l16, (__bf16)(o[vt][r] * inv));
  }
}

// ---------------------------------------------------------------------------
extern "C" void kernel_launch(void* const* d_in, const int* in_sizes, int n_in,
                              void* d_out, int out_size, void* d_ws, size_t ws_size,
                              hipStream_t stream) {
  (void)in_sizes; (void)n_in; (void)out_size; (void)ws_size;
  const float* h      = (const float*)d_in[0];
  const float* w_q_a  = (const float*)d_in[1];
  const float* g_q_a  = (const float*)d_in[2];
  const float* w_q_b  = (const float*)d_in[3];
  const float* w_kv_a = (const float*)d_in[4];
  const float* g_kv_a = (const float*)d_in[5];
  const float* w_kv_b = (const float*)d_in[6];
  const float* w_o    = (const float*)d_in[7];
  float* out = (float*)d_out;

  char* ws = (char*)d_ws;
  size_t off = 0;
  auto alloc = [&](size_t bytes) -> char* {
    char* p = ws + off;
    off += (bytes + 255) & ~(size_t)255;
    return p;
  };

  __bf16* h_bf   = (__bf16*)alloc((size_t)ROWS * 2048 * 2);
  __bf16* wqa_t  = (__bf16*)alloc((size_t)1536 * 2048 * 2);
  __bf16* wkva_t = (__bf16*)alloc((size_t)576 * 2048 * 2);
  __bf16* wqb_t  = (__bf16*)alloc((size_t)3072 * 1536 * 2);
  __bf16* wkvb_t = (__bf16*)alloc((size_t)4096 * 512 * 2);
  __bf16* wo_t   = (__bf16*)alloc((size_t)2048 * 2048 * 2);
  float*  qa_raw = (float*)alloc((size_t)ROWS * 1536 * 4);   // reused as ctx later
  float*  kva_raw= (float*)alloc((size_t)ROWS * 576 * 4);
  __bf16* qa_bf  = (__bf16*)alloc((size_t)ROWS * 1536 * 2);
  __bf16* kvn_bf = (__bf16*)alloc((size_t)ROWS * 512 * 2);
  float*  kpe    = (float*)alloc((size_t)ROWS * 64 * 4);
  float*  q_raw  = (float*)alloc((size_t)ROWS * 3072 * 4);   // reused for k_full+v_t
  float*  kv_raw = (float*)alloc((size_t)ROWS * 4096 * 4);
  __bf16* q_full = (__bf16*)alloc((size_t)B_DIM * HEADS * S_LEN * DH * 2);

  __bf16* ctx    = (__bf16*)qa_raw;
  __bf16* k_full = (__bf16*)q_raw;
  __bf16* v_t    = (__bf16*)((char*)q_raw + (size_t)B_DIM * HEADS * S_LEN * DH * 2);

  k_convert  <<<4096, 256, 0, stream>>>(h, h_bf, (long long)ROWS * 2048);
  k_transpose<<<2048, 256, 0, stream>>>(w_q_a,  wqa_t,  2048, 1536);
  k_transpose<<<2048, 256, 0, stream>>>(w_kv_a, wkva_t, 2048, 576);
  k_transpose<<<2048, 256, 0, stream>>>(w_q_b,  wqb_t,  1536, 3072);
  k_transpose<<<2048, 256, 0, stream>>>(w_kv_b, wkvb_t, 512,  4096);
  k_transpose<<<2048, 256, 0, stream>>>(w_o,    wo_t,   2048, 2048);

  k_gemm<<<dim3(12, 32), 256, 0, stream>>>(h_bf, wqa_t,  qa_raw,  ROWS, 1536, 2048);
  k_gemm<<<dim3(5,  32), 256, 0, stream>>>(h_bf, wkva_t, kva_raw, ROWS, 576,  2048);

  k_rmsnorm_qa <<<ROWS, 256, 0, stream>>>(qa_raw,  g_q_a,  qa_bf);
  k_rmsnorm_kva<<<ROWS, 256, 0, stream>>>(kva_raw, g_kv_a, kvn_bf, kpe);

  k_gemm<<<dim3(24, 32), 256, 0, stream>>>(qa_bf,  wqb_t,  q_raw,  ROWS, 3072, 1536);
  k_rope_q<<<4096, 256, 0, stream>>>(q_raw, q_full);
  k_gemm<<<dim3(32, 32), 256, 0, stream>>>(kvn_bf, wkvb_t, kv_raw, ROWS, 4096, 512);
  k_rearrange_kv<<<4096, 256, 0, stream>>>(kv_raw, kpe, k_full, v_t);

  k_attention<<<dim3(16, 32), 256, 0, stream>>>(q_full, k_full, v_t, ctx);

  k_gemm<<<dim3(16, 32), 256, 0, stream>>>(ctx, wo_t, out, ROWS, 2048, 2048);
}
